// TorchTreeNNModel_23132693856420
// MI455X (gfx1250) — matrix operand order repaired and checked by reading, not compiled, not run on hardware
//
#include <hip/hip_runtime.h>
#include <hip/hip_bf16.h>
#include <math.h>

typedef __attribute__((ext_vector_type(16))) _Float16 v16h;
typedef __attribute__((ext_vector_type(8)))  float    v8f;

#define Bv 512
#define Lv 512
#define Dv 128
#define ROW_H 136                            // halves per reps row (128 + 8 pad -> bank spread)
#define REPS_BYTES (Lv * ROW_H * 2)          // 139264 B
#define IDX_OFF    REPS_BYTES
#define LDS_BYTES  (REPS_BYTES + Lv * 3 * 4) // + 6144 B subtree indices = 145408 B

// Hardware tanh: single TRANS-unit op on CDNA5 instead of libm's exp-based
// expansion (which costs ~40 VALU ops + a divergent branch on the serial
// dependency chain of every tree step).
static __device__ __forceinline__ float fast_tanh(float x) {
#if __has_builtin(__builtin_amdgcn_tanhf)
  return __builtin_amdgcn_tanhf(x);
#else
  float y;
  asm("v_tanh_f32 %0, %1" : "=v"(y) : "v"(x));
  return y;
#endif
}

__global__ __launch_bounds__(256, 1)
void treenn_fused_kernel(const int* __restrict__ subtree,      // [B,L,3]
                         const int* __restrict__ lens,         // [B]
                         const int* __restrict__ emb_ind,      // [B,L]
                         const float* __restrict__ embedding,  // [V,D]
                         const float* __restrict__ W_tree,     // [D, 2D]
                         const float* __restrict__ b_tree,     // [D]
                         const float* __restrict__ W_cls,      // [3,D]
                         const float* __restrict__ b_cls,      // [3]
                         float* __restrict__ out)              // [B,3]
{
  extern __shared__ char smem[];
  _Float16* reps = (_Float16*)smem;              // [L][ROW_H] f16
  int*      sidx = (int*)(smem + IDX_OFF);       // [L*3]

  const int b    = blockIdx.x;
  const int tid  = threadIdx.x;
  const int lane = tid & 31;
  const int wave = tid >> 5;                     // M-tile id, 0..7

  // ---- stage subtree indices for this tree into LDS ----
  const int* sub = subtree + (size_t)b * Lv * 3;
  for (int j = tid; j < Lv * 3; j += 256) sidx[j] = sub[j];

  // ---- gather embedding rows -> LDS reps (f32 -> f16) ----
  const int* eind = emb_ind + (size_t)b * Lv;
  for (int task = tid; task < Lv * 8; task += 256) {
    const int row   = task >> 3;
    const int chunk = task & 7;                  // 16 floats per task
    const float* src = embedding + (size_t)eind[row] * Dv + chunk * 16;
    union { _Float16 h[16]; float4 f4[2]; } u;
    #pragma unroll
    for (int j = 0; j < 16; ++j) u.h[j] = (_Float16)src[j];
    float4* dst = (float4*)(reps + row * ROW_H + chunk * 16);
    dst[0] = u.f4[0];
    dst[1] = u.f4[1];
  }

  // ---- preload A fragments: this wave's 16 rows of W_tree, f16, 8 K-tiles ----
  // 16-bit A 16x32 layout: lanes 0-15 (M=lane) hold K {0..7,16..23}; lanes 16-31 hold K {8..15,24..31}
  const int rowA   = wave * 16 + (lane & 15);
  const int kbaseA = (lane < 16) ? 0 : 8;
  v16h afrag[8];
  #pragma unroll
  for (int kt = 0; kt < 8; ++kt) {
    const float* wrow = W_tree + (size_t)rowA * 256 + kt * 32 + kbaseA;
    #pragma unroll
    for (int j = 0; j < 16; ++j)
      afrag[kt][j] = (_Float16)wrow[j + (j & 8)];   // j<8 -> K j ; j>=8 -> K j+8
  }

  // C/D 16x16 f32 layout: lanes 0-15 rows v (0..7), lanes 16-31 rows 8+v; column = lane&15
  const int rowD0 = wave * 16 + ((lane < 16) ? 0 : 8);
  float bias[8];
  #pragma unroll
  for (int v = 0; v < 8; ++v) bias[v] = b_tree[rowD0 + v];

  const int n = lens[b];

  __syncthreads();

  // B 32x16 f16 layout: lanes 0-15 hold K {0..7,16..23}, lanes 16-31 hold K {8..15,24..31}
  const int koffB = (lane < 16) ? 0 : 8;

  // ---- sequential tree interpretation: 512 steps, 8 WMMAs per active step ----
  for (int i = 0; i < Lv; ++i) {
    const int p = sidx[i * 3 + 0];
    const int l = sidx[i * 3 + 1];
    const int r = sidx[i * 3 + 2];
    const bool doit = (l != r) && (i < n);        // uniform across the block
    if (doit) {
      __syncthreads();                            // prior step's reps[p] writes visible

      v16h bfrag[8];
      #pragma unroll
      for (int kt = 0; kt < 8; ++kt) {
        const int k0 = kt * 32 + koffB;           // 0..255 within combined [reps[l];reps[r]]
        const _Float16* srcrow = (k0 < 128) ? (reps + l * ROW_H + k0)
                                            : (reps + r * ROW_H + (k0 - 128));
        union { v16h v; float4 f4[2]; } u;
        u.f4[0] = ((const float4*)srcrow)[0];     // broadcast ds_load_b128
        u.f4[1] = ((const float4*)srcrow)[2];
        bfrag[kt] = u.v;
      }

      v8f c = {0.f, 0.f, 0.f, 0.f, 0.f, 0.f, 0.f, 0.f};
      #pragma unroll
      for (int kt = 0; kt < 8; ++kt)
        c = __builtin_amdgcn_wmma_f32_16x16x32_f16(false, afrag[kt], false, bfrag[kt],
                                                   (short)0, c, false, false);

      __syncthreads();                            // all reads done before overwriting reps[p]

      if ((lane & 15) == 0) {                     // lanes owning N-column 0 (the real tree)
        union { _Float16 h[8]; float4 f4; } o;
        #pragma unroll
        for (int v = 0; v < 8; ++v)
          o.h[v] = (_Float16)fast_tanh(c[v] + bias[v]);
        *(float4*)(reps + p * ROW_H + rowD0) = o.f4;   // ds_store_b128
      }
    }
  }

  __syncthreads();

  // ---- classifier: root = reps[0], out[b] = root @ W_cls^T + b_cls ----
  if (tid < 32) {
    float a0 = 0.f, a1 = 0.f, a2 = 0.f;
    for (int d = lane; d < Dv; d += 32) {
      const float rv = (float)reps[d];
      a0 += rv * W_cls[0 * Dv + d];
      a1 += rv * W_cls[1 * Dv + d];
      a2 += rv * W_cls[2 * Dv + d];
    }
    #pragma unroll
    for (int s = 16; s > 0; s >>= 1) {
      a0 += __shfl_xor(a0, s, 32);
      a1 += __shfl_xor(a1, s, 32);
      a2 += __shfl_xor(a2, s, 32);
    }
    if (lane == 0) {
      out[b * 3 + 0] = a0 + b_cls[0];
      out[b * 3 + 1] = a1 + b_cls[1];
      out[b * 3 + 2] = a2 + b_cls[2];
    }
  }
}

extern "C" void kernel_launch(void* const* d_in, const int* in_sizes, int n_in,
                              void* d_out, int out_size, void* d_ws, size_t ws_size,
                              hipStream_t stream) {
  const int*   subtree = (const int*)d_in[0];
  const int*   lens    = (const int*)d_in[1];
  const int*   eind    = (const int*)d_in[2];
  const float* emb     = (const float*)d_in[3];
  const float* Wt      = (const float*)d_in[4];
  const float* bt      = (const float*)d_in[5];
  const float* Wc      = (const float*)d_in[6];
  const float* bc      = (const float*)d_in[7];
  float* out = (float*)d_out;

  (void)in_sizes; (void)n_in; (void)out_size; (void)d_ws; (void)ws_size;

  // 145 KB dynamic LDS per workgroup (320 KB per WGP -> 2 trees per WGP)
  hipFuncSetAttribute((const void*)treenn_fused_kernel,
                      hipFuncAttributeMaxDynamicSharedMemorySize, LDS_BYTES);

  treenn_fused_kernel<<<dim3(Bv), dim3(256), LDS_BYTES, stream>>>(
      subtree, lens, eind, emb, Wt, bt, Wc, bc, out);
}
